// GraphAttentionLayer_25168508355325
// MI455X (gfx1250) — matrix operand order repaired and checked by reading, not compiled, not run on hardware
//
#include <hip/hip_runtime.h>
#include <cstring>

// Problem constants (match reference)
#define NN 50000
#define EE 800000
#define FIN 256
#define HH 4
#define DD 64
#define HD 256  // HH*DD
#define NEG_SLOPE 0.2f

typedef _Float16 v4h  __attribute__((ext_vector_type(4)));
typedef _Float16 v8h  __attribute__((ext_vector_type(8)));
typedef _Float16 v16h __attribute__((ext_vector_type(16)));
typedef float    v8f  __attribute__((ext_vector_type(8)));

__device__ __forceinline__ int f32_ord(float f) {
  int i = __float_as_int(f);
  return (i >= 0) ? i : (i ^ 0x7FFFFFFF);
}
__device__ __forceinline__ float ord_f32(int i) {
  return __int_as_float((i >= 0) ? i : (i ^ 0x7FFFFFFF));
}

// ---------------- utility kernels ----------------
__global__ void cast_f32_f16(const float* __restrict__ in, _Float16* __restrict__ out, int n) {
  int i = blockIdx.x * blockDim.x + threadIdx.x;
  if (i < n) out[i] = (_Float16)in[i];
}

// W[K, Ncols] row-major  ->  Wt[Ncols, K] row-major, f16
__global__ void transpose_cast(const float* __restrict__ W, _Float16* __restrict__ Wt,
                               int K, int Ncols) {
  int idx = blockIdx.x * blockDim.x + threadIdx.x;
  if (idx >= K * Ncols) return;
  int k = idx / Ncols, n = idx % Ncols;          // coalesced read of W
  Wt[(size_t)n * K + k] = (_Float16)W[idx];
}

__global__ void fill_i32(int* __restrict__ p, int v, int n) {
  int i = blockIdx.x * blockDim.x + threadIdx.x;
  if (i < n) p[i] = v;
}

// ---------------- WMMA GEMM: out[M,Ncols] = A16[M,256] * Wt[Ncols,256]^T + bias ----------------
// One wave -> 16x64 output tile.  K fixed = 256 (8 steps of 32).
// A layout per ISA 16-bit A 16x32: lanes 0-15 row M, halves K 0..7 then 16..23;
//                                  lanes 16-31 halves K 8..15 then 24..31.
// B (32x16, 16-bit): lane = column, same K order as A (lanes 0-15 K 0..7,16..23; lanes 16-31 K 8..15,24..31).
template <bool OUT16>
__global__ void gemm_wmma_16x64(const _Float16* __restrict__ A,
                                const _Float16* __restrict__ Wt,
                                const float* __restrict__ bias,
                                void* __restrict__ outP,
                                int M, int Ncols) {
  const int lane = threadIdx.x & 31;
  const int wave = threadIdx.x >> 5;
  const int ntn  = Ncols >> 6;                    // 64-col tile groups
  const int ntiles = (M >> 4) * ntn;
  const int tile = blockIdx.x * (blockDim.x >> 5) + wave;
  if (tile >= ntiles) return;                     // wave-uniform: EXEC stays all-ones
  const int mt = tile / ntn;
  const int nt = tile % ntn;
  const int m0 = mt << 4;
  const int n0 = nt << 6;
  const int r  = lane & 15;
  const int hi = lane >> 4;

  v8f acc0 = {}, acc1 = {}, acc2 = {}, acc3 = {};

  const _Float16* arow = A + (size_t)(m0 + r) * FIN + hi * 8;
  const size_t c0 = (size_t)(n0 + r) * FIN;
#pragma unroll
  for (int kk = 0; kk < FIN; kk += 32) {
    v8h alo = *(const v8h*)(arow + kk);
    v8h ahi = *(const v8h*)(arow + kk + 16);
    v16h a = __builtin_shufflevector(alo, ahi, 0,1,2,3,4,5,6,7,8,9,10,11,12,13,14,15);
    const _Float16* bb = Wt + kk + hi * 8;
    v16h b0 = __builtin_shufflevector(*(const v8h*)(bb + c0), *(const v8h*)(bb + c0 + 16), 0,1,2,3,4,5,6,7,8,9,10,11,12,13,14,15);
    v16h b1 = __builtin_shufflevector(*(const v8h*)(bb + c0 + 16 * FIN), *(const v8h*)(bb + c0 + 16 * FIN + 16), 0,1,2,3,4,5,6,7,8,9,10,11,12,13,14,15);
    v16h b2 = __builtin_shufflevector(*(const v8h*)(bb + c0 + 32 * FIN), *(const v8h*)(bb + c0 + 32 * FIN + 16), 0,1,2,3,4,5,6,7,8,9,10,11,12,13,14,15);
    v16h b3 = __builtin_shufflevector(*(const v8h*)(bb + c0 + 48 * FIN), *(const v8h*)(bb + c0 + 48 * FIN + 16), 0,1,2,3,4,5,6,7,8,9,10,11,12,13,14,15);
    acc0 = __builtin_amdgcn_wmma_f32_16x16x32_f16(false, a, false, b0, (short)0, acc0, false, false);
    acc1 = __builtin_amdgcn_wmma_f32_16x16x32_f16(false, a, false, b1, (short)0, acc1, false, false);
    acc2 = __builtin_amdgcn_wmma_f32_16x16x32_f16(false, a, false, b2, (short)0, acc2, false, false);
    acc3 = __builtin_amdgcn_wmma_f32_16x16x32_f16(false, a, false, b3, (short)0, acc3, false, false);
  }

  v8f accs[4] = {acc0, acc1, acc2, acc3};
#pragma unroll
  for (int t = 0; t < 4; ++t) {
    const int col = n0 + t * 16 + r;
    const float bv = bias[col];
#pragma unroll
    for (int i = 0; i < 8; ++i) {
      const int row = m0 + i + hi * 8;
      const float val = accs[t][i] + bv;
      if (OUT16) ((_Float16*)outP)[(size_t)row * Ncols + col] = (_Float16)val;
      else       ((float*)outP)[(size_t)row * Ncols + col] = val;
    }
  }
}

// ---------------- edge attention scores ----------------
// 16 lanes cooperate on one (edge, head) dot of length 64 (4 halves per lane).
__global__ void edge_scores(const _Float16* __restrict__ q16,
                            const _Float16* __restrict__ k16,
                            const int* __restrict__ ei,
                            float* __restrict__ scores) {
  const int gid  = blockIdx.x * blockDim.x + threadIdx.x;
  const int pair = gid >> 4;
  const int j    = gid & 15;
  if (pair >= EE * HH) return;
  const int h = pair & (HH - 1);
  const int e = pair >> 2;
  const int src = ei[e];
  const int dst = ei[EE + e];
  const v4h qa = *(const v4h*)(q16 + (size_t)src * HD + h * DD + j * 4);
  const v4h kb = *(const v4h*)(k16 + (size_t)dst * HD + h * DD + j * 4);
  float s = (float)qa[0] * (float)kb[0] + (float)qa[1] * (float)kb[1]
          + (float)qa[2] * (float)kb[2] + (float)qa[3] * (float)kb[3];
  s += __shfl_xor(s, 1, 32);
  s += __shfl_xor(s, 2, 32);
  s += __shfl_xor(s, 4, 32);
  s += __shfl_xor(s, 8, 32);
  if (j == 0) {
    s *= 0.125f;                                  // 1/sqrt(64)
    s = (s >= 0.f) ? s : NEG_SLOPE * s;           // LeakyReLU(0.2)
    scores[pair] = s;
  }
}

__global__ void seg_max(const float* __restrict__ scores, const int* __restrict__ ei,
                        int* __restrict__ m_ord) {
  const int p = blockIdx.x * blockDim.x + threadIdx.x;
  if (p >= EE * HH) return;
  const int h = p & (HH - 1);
  const int e = p >> 2;
  const int dst = ei[EE + e];
  atomicMax(&m_ord[dst * HH + h], f32_ord(scores[p]));
}

__global__ void seg_expsum(float* __restrict__ scores, const int* __restrict__ ei,
                           const int* __restrict__ m_ord, float* __restrict__ denom) {
  const int p = blockIdx.x * blockDim.x + threadIdx.x;
  if (p >= EE * HH) return;
  const int h = p & (HH - 1);
  const int e = p >> 2;
  const int dst = ei[EE + e];
  const float ex = __expf(scores[p] - ord_f32(m_ord[dst * HH + h]));
  scores[p] = ex;
  atomicAdd(&denom[dst * HH + h], ex);
}

// one block = one edge; thread = one of 256 channels
__global__ void scatter_v(const float* __restrict__ scores, const float* __restrict__ denom,
                          const _Float16* __restrict__ v16p, const int* __restrict__ ei,
                          float* __restrict__ accum) {
  const int e = blockIdx.x;
  const int c = threadIdx.x;
  const int src = ei[e];
  const int dst = ei[EE + e];
  const int h = c >> 6;
  const float a = scores[(size_t)e * HH + h] / denom[dst * HH + h];
  const float val = a * (float)v16p[(size_t)src * HD + c];
  atomicAdd(&accum[(size_t)dst * HD + c], val);
}

// ---------------- launcher ----------------
extern "C" void kernel_launch(void* const* d_in, const int* in_sizes, int n_in,
                              void* d_out, int out_size, void* d_ws, size_t ws_size,
                              hipStream_t stream) {
  const float* x  = (const float*)d_in[0];
  const int*   ei = (const int*)d_in[1];
  const float* Wq = (const float*)d_in[2];
  const float* bq = (const float*)d_in[3];
  const float* Wk = (const float*)d_in[4];
  const float* bk = (const float*)d_in[5];
  const float* Wv = (const float*)d_in[6];
  const float* bv = (const float*)d_in[7];
  const float* Wo = (const float*)d_in[8];
  const float* bo = (const float*)d_in[9];
  float* out = (float*)d_out;

  // workspace layout
  char* base = (char*)d_ws;
  size_t off = 0;
  auto take = [&](size_t bytes) -> char* {
    char* p = base + off;
    off = (off + bytes + 255) & ~(size_t)255;
    return p;
  };
  _Float16* x16   = (_Float16*)take((size_t)NN * FIN * 2);
  _Float16* wtq   = (_Float16*)take((size_t)HD * FIN * 2);
  _Float16* wtk   = (_Float16*)take((size_t)HD * FIN * 2);
  _Float16* wtv   = (_Float16*)take((size_t)HD * FIN * 2);
  _Float16* wto   = (_Float16*)take((size_t)DD * HD * 2);
  _Float16* q16   = (_Float16*)take((size_t)NN * HD * 2);
  _Float16* k16   = (_Float16*)take((size_t)NN * HD * 2);
  _Float16* v16p  = (_Float16*)take((size_t)NN * HD * 2);
  float*    scr   = (float*)take((size_t)EE * HH * 4);
  int*      m_ord = (int*)take((size_t)NN * HH * 4);
  float*    denom = (float*)take((size_t)NN * HH * 4);
  float*    accum = (float*)take((size_t)NN * HD * 4);
  _Float16* acc16 = (_Float16*)take((size_t)NN * HD * 2);
  (void)ws_size;

  const int BLK = 256;
  // 1) casts / transposes
  cast_f32_f16<<<(NN * FIN + BLK - 1) / BLK, BLK, 0, stream>>>(x, x16, NN * FIN);
  transpose_cast<<<(FIN * HD + BLK - 1) / BLK, BLK, 0, stream>>>(Wq, wtq, FIN, HD);
  transpose_cast<<<(FIN * HD + BLK - 1) / BLK, BLK, 0, stream>>>(Wk, wtk, FIN, HD);
  transpose_cast<<<(FIN * HD + BLK - 1) / BLK, BLK, 0, stream>>>(Wv, wtv, FIN, HD);
  transpose_cast<<<(FIN * DD + BLK - 1) / BLK, BLK, 0, stream>>>(Wo, wto, FIN, DD);

  // 2) init segment buffers
  float neg; int negbits;
  neg = -3.0e38f; memcpy(&negbits, &neg, 4);
  const int ordInit = (negbits >= 0) ? negbits : (negbits ^ 0x7FFFFFFF);
  fill_i32<<<(NN * HH + BLK - 1) / BLK, BLK, 0, stream>>>(m_ord, ordInit, NN * HH);
  hipMemsetAsync(denom, 0, (size_t)NN * HH * 4, stream);
  hipMemsetAsync(accum, 0, (size_t)NN * HD * 4, stream);

  // 3) QKV projections via WMMA (f16 outputs)
  const int qkvTiles = (NN / 16) * (HD / 64);
  const int qkvBlocks = (qkvTiles + 7) / 8;       // 8 waves per 256-thread block
  gemm_wmma_16x64<true><<<qkvBlocks, BLK, 0, stream>>>(x16, wtq, bq, q16, NN, HD);
  gemm_wmma_16x64<true><<<qkvBlocks, BLK, 0, stream>>>(x16, wtk, bk, k16, NN, HD);
  gemm_wmma_16x64<true><<<qkvBlocks, BLK, 0, stream>>>(x16, wtv, bv, v16p, NN, HD);

  // 4) edge scores (16 lanes per (e,h))
  const long long sThreads = (long long)EE * HH * 16;
  edge_scores<<<(unsigned)((sThreads + BLK - 1) / BLK), BLK, 0, stream>>>(q16, k16, ei, scr);

  // 5) segment softmax
  const int P = EE * HH;
  seg_max<<<(P + BLK - 1) / BLK, BLK, 0, stream>>>(scr, ei, m_ord);
  seg_expsum<<<(P + BLK - 1) / BLK, BLK, 0, stream>>>(scr, ei, m_ord, denom);

  // 6) weighted scatter to accum
  scatter_v<<<EE, BLK, 0, stream>>>(scr, denom, v16p, ei, accum);

  // 7) output projection via WMMA (f32 output)
  cast_f32_f16<<<(NN * HD + BLK - 1) / BLK, BLK, 0, stream>>>(accum, acc16, NN * HD);
  const int oTiles = (NN / 16) * (DD / 64);
  gemm_wmma_16x64<false><<<(oTiles + 7) / 8, BLK, 0, stream>>>(acc16, wto, bo, out, NN, DD);
}
